// DGCNN_26053271618159
// MI455X (gfx1250) — matrix operand order repaired and checked by reading, not compiled, not run on hardware
//
#include <hip/hip_runtime.h>

#define LEAK 0.2f
#define KNN  40
#define NPTS 4096
#define NB   4

typedef __attribute__((ext_vector_type(16))) _Float16 v16h;
typedef __attribute__((ext_vector_type(8)))  _Float16 v8h;
typedef __attribute__((ext_vector_type(8)))  float    v8f;

// leaky_relu(y) == max(y, LEAK*y) for 0 < LEAK < 1  -> v_max_num_f32
__device__ __forceinline__ float lrelu(float y){ return fmaxf(y, LEAK * y); }

// Order-independent float max via integer CAS (works on LDS and global).
__device__ __forceinline__ void atomicMaxF32(float* addr, float v){
  int* ia = (int*)addr;
  int old = __float_as_int(*addr);
  while (v > __int_as_float(old)){
    int assumed = old;
    old = atomicCAS(ia, assumed, __float_as_int(v));
    if (old == assumed) break;
  }
}

// A fragment (16x32 f16) from row-major LDS [M][lda]; two aligned 16B loads.
// Lanes 0-15: M=0..15, K= 0..7 & 16..23 ; lanes 16-31: same rows, K=8..15 & 24..31.
__device__ __forceinline__ v16h a_frag(const _Float16* A, int lda, int m0, int k0, int lane){
  const _Float16* p = A + (m0 + (lane & 15)) * lda + k0 + ((lane >> 4) << 3);
  v8h lo = *(const v8h*)p;
  v8h hi = *(const v8h*)(p + 16);
  return __builtin_shufflevector(lo, hi, 0,1,2,3,4,5,6,7,8,9,10,11,12,13,14,15);
}

// B stored fragment-swizzled in LDS: [frag][lane][16 halves] -> one 32B contiguous read.
__device__ __forceinline__ v16h b_frag_sw(const _Float16* Bsw, int frag, int lane){
  return *(const v16h*)(Bsw + (((frag << 5) + lane) << 4));
}

// Swizzle index for element (k, o) of a [Kt x 64] W^T tile (Kt multiple of 32):
// frag = (k/32)*4 + o/16 ; lane = (o&15) + 16*((k>>3)&1) ; h = (k&7) + 8*((k>>4)&1).
__device__ __forceinline__ int b_sw_index(int k, int o){
  int frag = ((k >> 5) << 2) + (o >> 4);
  int lane = (o & 15) | ((k & 8) << 1);
  return ((frag << 5) + lane) * 16 + ((k & 7) | ((k >> 1) & 8));
}

__device__ __forceinline__ v8f wmma_f16(v16h a, v16h b, v8f c){
  return __builtin_amdgcn_wmma_f32_16x16x32_f16(false, a, false, b, (short)0, c, false, false);
}

// ---------------------------------------------------------------- transpose (B,3,N) -> (B,N,3)
__global__ void transpose_kernel(const float* __restrict__ x, float* __restrict__ xt){
  int t = blockIdx.x * blockDim.x + threadIdx.x;
  if (t >= NB * NPTS) return;
  int b = t >> 12, n = t & (NPTS - 1);
#pragma unroll
  for (int c = 0; c < 3; ++c)
    xt[(size_t)t * 3 + c] = x[(size_t)b * 3 * NPTS + c * NPTS + n];
}

// ---------------------------------------------------------------- kNN (top-40 of -||xi-xj||^2)
__global__ void knn_kernel(const float* __restrict__ x, int* __restrict__ idx){
  int t = blockIdx.x * blockDim.x + threadIdx.x;
  if (t >= NB * NPTS) return;
  int b = t >> 12, n = t & (NPTS - 1);
  const float* xb = x + (size_t)b * 3 * NPTS;
  float x0 = xb[n], x1 = xb[NPTS + n], x2 = xb[2 * NPTS + n];
  float vals[KNN]; int ids[KNN];
#pragma unroll
  for (int k = 0; k < KNN; ++k){ vals[k] = -3.0e38f; ids[k] = 0; }
  for (int j = 0; j < NPTS; ++j){
    float d0 = xb[j] - x0, d1 = xb[NPTS + j] - x1, d2 = xb[2 * NPTS + j] - x2;
    float pd = -(d0 * d0 + d1 * d1 + d2 * d2);
    if (pd > vals[KNN - 1]){
      int pos = KNN - 1;
      while (pos > 0 && vals[pos - 1] < pd){
        vals[pos] = vals[pos - 1]; ids[pos] = ids[pos - 1]; --pos;
      }
      vals[pos] = pd; ids[pos] = j;
    }
  }
  for (int k = 0; k < KNN; ++k) idx[(size_t)t * KNN + k] = ids[k];
}

// ---------------------------------------------------------------- fused edge-conv stage
// One block = 2 points x 40 neighbors = 80 rows. GEMM 80x64xKD (+ optional 80x64x64),
// leaky-relu, max over K, write (B,N,64). Edge features live entirely in LDS.
template<int CIN, int KD, bool TWO>
__global__ __launch_bounds__(256)
void edgeconv_kernel(const float* __restrict__ xin, const int* __restrict__ idx,
                     const float* __restrict__ Wa, const float* __restrict__ sa, const float* __restrict__ ba,
                     const float* __restrict__ Wb, const float* __restrict__ sb, const float* __restrict__ bb,
                     float* __restrict__ xout)
{
  constexpr int M = 2 * KNN;  // 80 rows
  __shared__ __attribute__((aligned(32))) _Float16 ebuf[M * KD];
  __shared__ __attribute__((aligned(32))) _Float16 wa[KD * 64];
  __shared__ __attribute__((aligned(32))) _Float16 wb[TWO ? 64 * 64 : 32];
  __shared__ __attribute__((aligned(32))) _Float16 c1 [TWO ? M * 64 : 32];
  __shared__ float    mx[2 * 64];
  __shared__ float    ctrf[2 * CIN];
  __shared__ int      jrow[M];

  const int tid  = threadIdx.x;
  const int lane = tid & 31;
  const int wave = tid >> 5;
  const int blk  = blockIdx.x;
  const int b    = blk / (NPTS / 2);
  const int pbase = (blk % (NPTS / 2)) * 2;

  // Stage weights as f16, fragment-swizzled; zero-pad K beyond 2*CIN.
  for (int i = tid; i < KD * 64; i += 256){
    int k = i >> 6, o = i & 63;
    wa[b_sw_index(k, o)] = (k < 2 * CIN) ? (_Float16)Wa[o * (2 * CIN) + k] : (_Float16)0.f;
  }
  if (TWO)
    for (int i = tid; i < 64 * 64; i += 256){
      int k = i >> 6, o = i & 63;
      wb[b_sw_index(k, o)] = (_Float16)Wb[o * 64 + k];
    }
  if (tid < M){
    int p = tid / KNN, kk = tid % KNN;
    jrow[tid] = idx[((size_t)b * NPTS + pbase + p) * KNN + kk];
  }
  if (tid < 2 * CIN)
    ctrf[tid] = xin[((size_t)b * NPTS + pbase + tid / CIN) * CIN + (tid % CIN)];
  if (tid < 128) mx[tid] = -3.0e38f;
  __syncthreads();

  // Edge features: [nbr - ctr | ctr], zero-padded to KD, f16.
  for (int i = tid; i < M * KD; i += 256){
    int e = i / KD, kf = i % KD;
    int p = e / KNN;
    float v = 0.f;
    if (kf < CIN){
      int j = jrow[e];
      v = xin[((size_t)b * NPTS + j) * CIN + kf] - ctrf[p * CIN + kf];
    } else if (kf < 2 * CIN){
      v = ctrf[p * CIN + (kf - CIN)];
    }
    ebuf[e * KD + kf] = (_Float16)v;
  }
  __syncthreads();

  // conv1: 5 M-tiles x 4 N-tiles = 20 WMMA tiles over 8 waves; fused epilogue.
#pragma unroll
  for (int ti = 0; ti < 3; ++ti){
    const int t = wave + ti * 8;
    if (t < 20){
      const int mt = t >> 2, nt = t & 3;
      v8f c = {};
#pragma unroll
      for (int ks = 0; ks < KD; ks += 32)
        c = wmma_f16(a_frag(ebuf, KD, mt * 16, ks, lane),
                     b_frag_sw(wa, (ks >> 5) * 4 + nt, lane), c);
      const int ch = nt * 16 + (lane & 15);
      const float sc = sa[ch], bc = ba[ch];
#pragma unroll
      for (int r = 0; r < 8; ++r){
        int m = mt * 16 + r + ((lane >> 4) << 3);
        float y = lrelu(c[r] * sc + bc);
        if (TWO) c1[m * 64 + ch] = (_Float16)y;
        else     atomicMaxF32(&mx[(m / KNN) * 64 + ch], y);
      }
    }
  }

  if (TWO){
    __syncthreads();
#pragma unroll
    for (int ti = 0; ti < 3; ++ti){
      const int t = wave + ti * 8;
      if (t < 20){
        const int mt = t >> 2, nt = t & 3;
        v8f c = {};
#pragma unroll
        for (int ks = 0; ks < 64; ks += 32)
          c = wmma_f16(a_frag(c1, 64, mt * 16, ks, lane),
                       b_frag_sw(wb, (ks >> 5) * 4 + nt, lane), c);
        const int ch = nt * 16 + (lane & 15);
        const float sc = sb[ch], bc = bb[ch];
#pragma unroll
        for (int r = 0; r < 8; ++r){
          int m = mt * 16 + r + ((lane >> 4) << 3);
          atomicMaxF32(&mx[(m / KNN) * 64 + ch], lrelu(c[r] * sc + bc));
        }
      }
    }
  }
  __syncthreads();
  if (tid < 128){
    int p = tid >> 6, ch = tid & 63;
    xout[((size_t)b * NPTS + pbase + p) * 64 + ch] = mx[tid];
  }
}

// ---------------------------------------------------------------- generic tiled WMMA GEMM
// Block tile 32 rows x 64 cols, K staged in 64-wide LDS chunks, 8 waves = 2x4 WMMA tiles.
// MODE 0: A = [x1|x2|x3] (K=192), epilogue lrelu + per-block column max -> pmax
// MODE 1: A = [gmax(b)|x1|x2|x3] (K=1216), epilogue lrelu -> f16 out
// MODE 2: A = f16 matrix (ws),             epilogue lrelu -> f16 out
// MODE 3: A = f16 matrix (ws), epilogue +bias, transposed (B,63,N) f32 store
// Modes 2/3 stage A via GLOBAL_LOAD_ASYNC_TO_LDS_B128 (ASYNCcnt / s_wait_asynccnt).
template<int MODE>
__global__ __launch_bounds__(256)
void gemm_kernel(const float* __restrict__ A0, const float* __restrict__ A1,
                 const float* __restrict__ A2, const float* __restrict__ A3,
                 const _Float16* __restrict__ Ah,
                 const float* __restrict__ W, const float* __restrict__ s, const float* __restrict__ bias,
                 _Float16* __restrict__ outh, float* __restrict__ outf,
                 int Kdim, int Ncols, int Oreal)
{
  __shared__ __attribute__((aligned(32))) _Float16 abuf[32 * 64];
  __shared__ __attribute__((aligned(32))) _Float16 bbuf[64 * 64];
  __shared__ float pmx[64];

  const int tid = threadIdx.x, lane = tid & 31, wave = tid >> 5;
  const int row0 = blockIdx.x * 32;
  const int col0 = blockIdx.y * 64;
  if (MODE == 0 && tid < 64) pmx[tid] = -3.0e38f;

  const int mt = wave >> 2, nt = wave & 3;
  v8f c = {};
  for (int kc = 0; kc < Kdim; kc += 64){
    __syncthreads();
    if (MODE >= 2){
      // Async DMA of the 32x64 f16 A tile: 256 threads x one 16B chunk, no VGPR round-trip.
      int r = tid >> 3, q = tid & 7;
      unsigned ldsoff = (unsigned)(uintptr_t)&abuf[r * 64 + q * 8];
      unsigned voff   = (unsigned)(((size_t)(row0 + r) * Kdim + kc + q * 8) * sizeof(_Float16));
      unsigned long long gbase = (unsigned long long)(uintptr_t)Ah;
      asm volatile("global_load_async_to_lds_b128 %0, %1, %2"
                   :: "v"(ldsoff), "v"(voff), "s"(gbase) : "memory");
    } else {
      for (int i = tid; i < 32 * 64; i += 256){
        int r = i >> 6, kk = kc + (i & 63);
        int row = row0 + r;
        if (MODE == 0){
          float v = (kk < 64)  ? A1[(size_t)row * 64 + kk]
                  : (kk < 128) ? A2[(size_t)row * 64 + kk - 64]
                               : A3[(size_t)row * 64 + kk - 128];
          abuf[i] = (_Float16)v;
        } else { // MODE 1
          int bb = row >> 12;
          float v = (kk < 1024) ? A0[bb * 1024 + kk]
                  : (kk < 1088) ? A1[(size_t)row * 64 + kk - 1024]
                  : (kk < 1152) ? A2[(size_t)row * 64 + kk - 1088]
                                : A3[(size_t)row * 64 + kk - 1152];
          abuf[i] = (_Float16)v;
        }
      }
    }
    for (int i = tid; i < 64 * 64; i += 256){
      int k = i >> 6, o = i & 63;
      int kg = kc + k, og = col0 + o;
      const float* wp = W + (size_t)og * Kdim + kg;
      if (kc + 64 < Kdim) __builtin_prefetch(wp + 64, 0, 0);   // next K-chunk of this row
      bbuf[b_sw_index(k, o)] = (og < Oreal) ? (_Float16)*wp : (_Float16)0.f;
    }
    if (MODE >= 2) asm volatile("s_wait_asynccnt 0x0" ::: "memory");
    __syncthreads();
#pragma unroll
    for (int ks = 0; ks < 64; ks += 32)
      c = wmma_f16(a_frag(abuf, 64, mt * 16, ks, lane),
                   b_frag_sw(bbuf, (ks >> 5) * 4 + nt, lane), c);
  }

  const int col = col0 + nt * 16 + (lane & 15);
#pragma unroll
  for (int r = 0; r < 8; ++r){
    int row = row0 + mt * 16 + r + ((lane >> 4) << 3);
    if (MODE == 3){
      if (col < Oreal){
        int bb = row >> 12, n = row & (NPTS - 1);
        outf[((size_t)bb * Oreal + col) * NPTS + n] = c[r] + bias[col];
      }
    } else {
      float y = lrelu(c[r] * s[col] + bias[col]);
      if (MODE == 0) atomicMaxF32(&pmx[nt * 16 + (lane & 15)], y);
      else           outh[(size_t)row * Ncols + col] = (_Float16)y;
    }
  }
  if (MODE == 0){
    __syncthreads();
    if (tid < 64) outf[(size_t)blockIdx.x * 1024 + col0 + tid] = pmx[tid];
  }
}

// ---------------------------------------------------------------- reduce per-block maxima to gmax (B,1024)
__global__ void gmax_reduce_kernel(const float* __restrict__ pmax, float* __restrict__ gmax){
  int t = blockIdx.x * blockDim.x + threadIdx.x;
  if (t >= NB * 1024) return;
  int b = t >> 10, ch = t & 1023;
  float m = -3.0e38f;
  for (int rb = 0; rb < 128; ++rb)
    m = fmaxf(m, pmax[((size_t)(b * 128 + rb)) * 1024 + ch]);
  gmax[t] = m;
}

// ---------------------------------------------------------------- launch
extern "C" void kernel_launch(void* const* d_in, const int* in_sizes, int n_in,
                              void* d_out, int out_size, void* d_ws, size_t ws_size,
                              hipStream_t stream)
{
  const float* x  = (const float*)d_in[0];
  const float* W1 = (const float*)d_in[1];  const float* s1 = (const float*)d_in[2];  const float* b1 = (const float*)d_in[3];
  const float* W2 = (const float*)d_in[4];  const float* s2 = (const float*)d_in[5];  const float* b2 = (const float*)d_in[6];
  const float* W3 = (const float*)d_in[7];  const float* s3 = (const float*)d_in[8];  const float* b3 = (const float*)d_in[9];
  const float* W4 = (const float*)d_in[10]; const float* s4 = (const float*)d_in[11]; const float* b4 = (const float*)d_in[12];
  const float* W5 = (const float*)d_in[13]; const float* s5 = (const float*)d_in[14]; const float* b5 = (const float*)d_in[15];
  const float* W6 = (const float*)d_in[16]; const float* s6 = (const float*)d_in[17]; const float* b6 = (const float*)d_in[18];
  const float* W7 = (const float*)d_in[19]; const float* s7 = (const float*)d_in[20]; const float* b7 = (const float*)d_in[21];
  const float* W8 = (const float*)d_in[22]; const float* s8 = (const float*)d_in[23]; const float* b8 = (const float*)d_in[24];
  const float* W9 = (const float*)d_in[25]; const float* b9 = (const float*)d_in[26];

  char* wp = (char*)d_ws;
  auto alloc = [&](size_t bytes) -> void* {
    void* p = (void*)wp;
    wp += (bytes + 255) & ~(size_t)255;
    return p;
  };
  float*    xt   = (float*)alloc((size_t)NB * NPTS * 3 * sizeof(float));
  int*      idx  = (int*)  alloc((size_t)NB * NPTS * KNN * sizeof(int));
  float*    x1   = (float*)alloc((size_t)NB * NPTS * 64 * sizeof(float));
  float*    x2   = (float*)alloc((size_t)NB * NPTS * 64 * sizeof(float));
  float*    x3   = (float*)alloc((size_t)NB * NPTS * 64 * sizeof(float));
  float*    pmax = (float*)alloc((size_t)(NB * NPTS / 32) * 1024 * sizeof(float));
  float*    gmax = (float*)alloc((size_t)NB * 1024 * sizeof(float));
  _Float16* h7   = (_Float16*)alloc((size_t)NB * NPTS * 512 * sizeof(_Float16));
  _Float16* h8   = (_Float16*)alloc((size_t)NB * NPTS * 256 * sizeof(_Float16));

  const int rowBlocks = NB * NPTS / 32;   // 512

  transpose_kernel<<<NB * NPTS / 256, 256, 0, stream>>>(x, xt);
  knn_kernel<<<NB * NPTS / 256, 256, 0, stream>>>(x, idx);

  edgeconv_kernel<3, 32, true><<<NB * NPTS / 2, 256, 0, stream>>>(
      xt, idx, W1, s1, b1, W2, s2, b2, x1);
  edgeconv_kernel<64, 128, true><<<NB * NPTS / 2, 256, 0, stream>>>(
      x1, idx, W3, s3, b3, W4, s4, b4, x2);
  edgeconv_kernel<64, 128, false><<<NB * NPTS / 2, 256, 0, stream>>>(
      x2, idx, W5, s5, b5, nullptr, nullptr, nullptr, x3);

  gemm_kernel<0><<<dim3(rowBlocks, 16), 256, 0, stream>>>(
      nullptr, x1, x2, x3, nullptr, W6, s6, b6, nullptr, pmax, 192, 1024, 1024);
  gmax_reduce_kernel<<<(NB * 1024 + 255) / 256, 256, 0, stream>>>(pmax, gmax);

  gemm_kernel<1><<<dim3(rowBlocks, 8), 256, 0, stream>>>(
      gmax, x1, x2, x3, nullptr, W7, s7, b7, h7, nullptr, 1216, 512, 512);
  gemm_kernel<2><<<dim3(rowBlocks, 4), 256, 0, stream>>>(
      nullptr, nullptr, nullptr, nullptr, h7, W8, s8, b8, h8, nullptr, 512, 256, 256);
  gemm_kernel<3><<<dim3(rowBlocks, 1), 256, 0, stream>>>(
      nullptr, nullptr, nullptr, nullptr, h8, W9, nullptr, b9, nullptr, (float*)d_out, 256, 64, 63);
}
